// FILM_84086869721201
// MI455X (gfx1250) — matrix operand order repaired and checked by reading, not compiled, not run on hardware
//
#include <hip/hip_runtime.h>
#include <stdint.h>

// ---------------- problem constants (from reference) ----------------
#define N_NODES 100000   // multiple of 16 (6250 tiles)
#define N_EDGES 200000
#define NFEAT   14
#define K0PAD   32       // layer-0 K padded 14 -> 32
#define HID     320      // multiple of 32
#define DIMOUT  16
#define EPSV    1e-5f

typedef __attribute__((ext_vector_type(16))) __bf16 v16bf;
typedef __attribute__((ext_vector_type(8)))  __bf16 v8bf;
typedef __attribute__((ext_vector_type(8)))  float  v8f;

static inline int cdiv(long long a, long long b) { return (int)((a + b - 1) / b); }

// ---------------------------------------------------------------
// Degree (segment count) — computed once, reused by all 4 layers.
__global__ void k_degree(const long long* __restrict__ dst, float* __restrict__ cnt, int E) {
    int gid = blockIdx.x * blockDim.x + threadIdx.x;
    if (gid < E) atomicAdd(&cnt[(int)dst[gid]], 1.0f);
}

// Pack x (N x 14 f32) -> X0 (N x 32 bf16, zero padded K)
__global__ void k_pack_x(const float* __restrict__ x, __bf16* __restrict__ X0) {
    int gid = blockIdx.x * blockDim.x + threadIdx.x;
    if (gid >= N_NODES * K0PAD) return;
    int r = gid >> 5, j = gid & 31;
    float v = (j < NFEAT) ? x[r * NFEAT + j] : 0.0f;
    X0[gid] = (__bf16)v;
}

// Pack [Wl | Wf | Ws | Wfs] (din x 6*dout f32, K zero-padded to dinp) into the
// per-lane B fragments for V_WMMA_F32_16X16X32_BF16, in this file's contiguous k order (same as A):
//   tile (tk,tc): lane = 16*(kk>=16) + (c&15), element e = kk&15  (VGPR j holds K=2j,2j+1)
// One 16x32 B tile = 32 lanes * 16 bf16 = 1024 bytes, contiguous.
__global__ void k_pack_w(const float* __restrict__ Wl, const float* __restrict__ Wf,
                         const float* __restrict__ Ws, const float* __restrict__ Wfs,
                         __bf16* __restrict__ Bp, int din, int dinp, int dout) {
    int ncols = 6 * dout;
    int total = dinp * ncols;
    int gid = blockIdx.x * blockDim.x + threadIdx.x;
    if (gid >= total) return;
    int k = gid / ncols;
    int c = gid - k * ncols;
    float v = 0.0f;
    if (k < din) {
        if (c < dout)            v = Wl [k * dout     + c];
        else if (c < 3 * dout)   v = Wf [k * 2 * dout + (c - dout)];
        else if (c < 4 * dout)   v = Ws [k * dout     + (c - 3 * dout)];
        else                     v = Wfs[k * 2 * dout + (c - 4 * dout)];
    }
    int tk = k >> 5, kk = k & 31;
    int tc = c >> 4, cc = c & 15;
    int lane = ((kk >> 4) << 4) + cc;
    int e    = kk & 15;
    int nct  = ncols >> 4;
    Bp[(((size_t)tk * nct + tc) * 32 + lane) * 16 + e] = (__bf16)v;
}

// ---------------------------------------------------------------
// bf16 WMMA GEMM with LDS-staged B panel and double-buffered A fragments.
//   Block = 256 threads = 8 waves. Block -> 8 M-tiles x (CT*16) columns.
//   B panel (KT K-steps x CT tiles, pre-swizzled fragments) is copied
//   global->LDS once per block with global_load_async_to_lds_b128, then each
//   wave streams it via ds_load_b128 while A is double-buffered from global.
// Epilogue scatters each 16-col tile into compact slices of outBase:
//   seg0=h, seg1=beta(+bias), seg2=gamma(+bias), seg3=xs, seg4=beta_s, seg5=gamma_s
template <int CT, int KT>
__global__ __launch_bounds__(256) void k_gemm(const __bf16* __restrict__ A, int lda,
                                              const __bf16* __restrict__ Bp,
                                              float* __restrict__ outBase,
                                              const float* __restrict__ bias,
                                              int nct, int dout) {
    __shared__ __align__(16) char smem[KT * CT * 1024];

    const int mtiles  = N_NODES / 16;
    const int cgroups = nct / CT;
    const int mb = blockIdx.x / cgroups;       // 8-M-tile group
    const int cg = blockIdx.x - mb * cgroups;  // column group
    const int lane = threadIdx.x & 31;
    const int wib  = threadIdx.x >> 5;         // wave in block -> M tile
    const int mt   = mb * 8 + wib;

    // ---- async stage B panel (KT*CT tiles of 1KB) into LDS ----
    {
        const int totalChunks = KT * CT * 64;              // 16B chunks
        for (int c = threadIdx.x; c < totalChunks; c += 256) {
            int tileIdx = c >> 6;                          // kt*CT + t
            int within  = c & 63;
            int kt = tileIdx / CT;
            int t  = tileIdx - kt * CT;
            const char* src = (const char*)Bp +
                ((size_t)(kt * nct + cg * CT + t)) * 1024 + within * 16;
            unsigned ldsoff = (unsigned)(uintptr_t)(smem + tileIdx * 1024 + within * 16);
            asm volatile("global_load_async_to_lds_b128 %0, %1, off"
                         :: "v"(ldsoff), "v"((unsigned long long)(uintptr_t)src)
                         : "memory");
        }
        asm volatile("s_wait_asynccnt 0x0" ::: "memory");
        __syncthreads();
    }

    if (mt >= mtiles) return;

    const int hi  = lane >> 4;                 // half-wave select
    const int row = mt * 16 + (lane & 15);
    // A fragment (16x32 bf16): lane<16 holds K {0..15}, lane>=16 K {16..31}
    const __bf16* arow = A + (size_t)row * lda + hi * 16;

    v8f acc[CT];
#pragma unroll
    for (int t = 0; t < CT; ++t) acc[t] = (v8f){0.f,0.f,0.f,0.f,0.f,0.f,0.f,0.f};

    union AB { v16bf v; v8bf h[2]; };
    AB a_cur, a_nxt;
    a_cur.h[0] = *(const v8bf*)(arow);
    a_cur.h[1] = *(const v8bf*)(arow + 8);

#pragma unroll
    for (int kt = 0; kt < KT; ++kt) {
        if (kt + 1 < KT) {                     // double-buffer next A fragment
            a_nxt.h[0] = *(const v8bf*)(arow + (kt + 1) * 32);
            a_nxt.h[1] = *(const v8bf*)(arow + (kt + 1) * 32 + 8);
        }
#pragma unroll
        for (int t = 0; t < CT; ++t) {
            const char* bl = smem + ((kt * CT + t) * 1024 + lane * 32);
            AB b;
            b.h[0] = *(const v8bf*)(bl);       // ds_load_b128
            b.h[1] = *(const v8bf*)(bl + 16);
            acc[t] = __builtin_amdgcn_wmma_f32_16x16x32_bf16(
                         false, a_cur.v, false, b.v, (short)0, acc[t], false, false);
        }
        a_cur = a_nxt;
    }

    // C/D layout: VGPR v, lane l -> row = v + 8*(l>=16), col = l&15
    const int rbase = mt * 16 + hi * 8;
#pragma unroll
    for (int t = 0; t < CT; ++t) {
        int c   = (cg * CT + t) * 16 + (lane & 15);
        int seg = c / dout;                    // uniform per 16-col tile
        int j   = c - seg * dout;
        float badd = (seg == 1) ? bias[j] : (seg == 2 ? bias[dout + j] : 0.0f);
        float* dst = outBase + (size_t)seg * N_NODES * dout;
#pragma unroll
        for (int v = 0; v < 8; ++v)
            dst[(size_t)(rbase + v) * dout + j] = acc[t][v] + badd;
    }
}

// ---------------------------------------------------------------
// Edge FiLM message + scatter-add:  agg[dst] += relu?(gamma[dst]*h[src] + beta[dst])
__global__ void k_edge(const long long* __restrict__ src, const long long* __restrict__ dst,
                       const float* __restrict__ base, float* __restrict__ agg,
                       int dout, int act) {
    int per = dout >> 2;
    long long gid = (long long)blockIdx.x * blockDim.x + threadIdx.x;
    if (gid >= (long long)N_EDGES * per) return;
    int e  = (int)(gid / per);
    int j4 = (int)(gid - (long long)e * per) * 4;
    int s = (int)src[e], d = (int)dst[e];
    const float* h     = base;
    const float* beta  = base + 1 * (size_t)N_NODES * dout;
    const float* gamma = base + 2 * (size_t)N_NODES * dout;
    float4 hv = *(const float4*)(h     + (size_t)s * dout + j4);
    float4 gv = *(const float4*)(gamma + (size_t)d * dout + j4);
    float4 bv = *(const float4*)(beta  + (size_t)d * dout + j4);
    float m0 = gv.x * hv.x + bv.x;
    float m1 = gv.y * hv.y + bv.y;
    float m2 = gv.z * hv.z + bv.z;
    float m3 = gv.w * hv.w + bv.w;
    if (act) { m0 = fmaxf(m0, 0.f); m1 = fmaxf(m1, 0.f); m2 = fmaxf(m2, 0.f); m3 = fmaxf(m3, 0.f); }
    float* a = agg + (size_t)d * dout + j4;
    atomicAdd(a + 0, m0); atomicAdd(a + 1, m1);
    atomicAdd(a + 2, m2); atomicAdd(a + 3, m3);
}

// FiLM-skip self path + mean aggregation:  F = relu?(gs*xs+bs) + agg/max(cnt,1)
__global__ void k_self_agg(const float* __restrict__ base, const float* __restrict__ agg,
                           const float* __restrict__ cnt, float* __restrict__ F,
                           int dout, int act) {
    long long gid = (long long)blockIdx.x * blockDim.x + threadIdx.x;
    if (gid >= (long long)N_NODES * dout) return;
    int n = (int)(gid / dout);
    const float* xs = base + 3 * (size_t)N_NODES * dout;
    const float* bs = base + 4 * (size_t)N_NODES * dout;
    const float* gs = base + 5 * (size_t)N_NODES * dout;
    float self = gs[gid] * xs[gid] + bs[gid];
    if (act) self = fmaxf(self, 0.f);
    float inv = 1.0f / fmaxf(cnt[n], 1.0f);
    F[gid] = self + agg[gid] * inv;
}

// Column statistics for batchnorm (block-local accumulation, few atomics)
__global__ void k_colstats(const float* __restrict__ F, float* __restrict__ sum,
                           float* __restrict__ sq, int dout) {
    int j = blockIdx.y * blockDim.x + threadIdx.x;
    if (j >= dout) return;
    int r0 = blockIdx.x * 512;
    int r1 = r0 + 512; if (r1 > N_NODES) r1 = N_NODES;
    float s = 0.f, ss = 0.f;
    for (int r = r0; r < r1; ++r) {
        float v = F[(size_t)r * dout + j];
        s += v; ss += v * v;
    }
    atomicAdd(&sum[j], s);
    atomicAdd(&sq[j], ss);
}

// Batchnorm apply + convert to bf16 for next layer's A matrix
__global__ void k_bn_apply(const float* __restrict__ F, const float* __restrict__ sum,
                           const float* __restrict__ sq, const float* __restrict__ g,
                           const float* __restrict__ b, __bf16* __restrict__ Hb, int dout) {
    long long gid = (long long)blockIdx.x * blockDim.x + threadIdx.x;
    if (gid >= (long long)N_NODES * dout) return;
    int j = (int)(gid % dout);
    const float invN = 1.0f / (float)N_NODES;
    float mu  = sum[j] * invN;
    float var = sq[j] * invN - mu * mu;
    float v = (F[gid] - mu) * rsqrtf(var + EPSV) * g[j] + b[j];
    Hb[gid] = (__bf16)v;
}

// ---------------------------------------------------------------
extern "C" void kernel_launch(void* const* d_in, const int* in_sizes, int n_in,
                              void* d_out, int out_size, void* d_ws, size_t ws_size,
                              hipStream_t stream) {
    (void)in_sizes; (void)n_in; (void)out_size; (void)ws_size;
    const float*     x  = (const float*)d_in[0];
    const long long* ei = (const long long*)d_in[1];
    const float* lin_w[4]   = {(const float*)d_in[2],  (const float*)d_in[3],  (const float*)d_in[4],  (const float*)d_in[5]};
    const float* film_w[4]  = {(const float*)d_in[6],  (const float*)d_in[7],  (const float*)d_in[8],  (const float*)d_in[9]};
    const float* film_b[4]  = {(const float*)d_in[10], (const float*)d_in[11], (const float*)d_in[12], (const float*)d_in[13]};
    const float* skip_w[4]  = {(const float*)d_in[14], (const float*)d_in[15], (const float*)d_in[16], (const float*)d_in[17]};
    const float* fskip_w[4] = {(const float*)d_in[18], (const float*)d_in[19], (const float*)d_in[20], (const float*)d_in[21]};
    const float* bn_g[3]    = {(const float*)d_in[22], (const float*)d_in[23], (const float*)d_in[24]};
    const float* bn_b[3]    = {(const float*)d_in[25], (const float*)d_in[26], (const float*)d_in[27]};

    // ---- workspace carve (total ~1.1 GB) ----
    uintptr_t p = (uintptr_t)d_ws;
    auto carve = [&](size_t bytes) -> void* {
        void* r = (void*)p; p += (bytes + 255) & ~(size_t)255; return r;
    };
    __bf16* Bp[4];
    const size_t bpel[4] = {(size_t)K0PAD * 6 * HID, (size_t)HID * 6 * HID,
                            (size_t)HID * 6 * HID,  (size_t)HID * 6 * DIMOUT};
    for (int i = 0; i < 4; ++i) Bp[i] = (__bf16*)carve(bpel[i] * 2);
    __bf16* X0  = (__bf16*)carve((size_t)N_NODES * K0PAD * 2);
    __bf16* Hb  = (__bf16*)carve((size_t)N_NODES * HID * 2);
    float*  Yc  = (float*)carve(6ull * N_NODES * HID * 4);
    float*  agg = (float*)carve((size_t)N_NODES * HID * 4);
    float*  F   = (float*)carve((size_t)N_NODES * HID * 4);
    float*  cnt = (float*)carve((size_t)N_NODES * 4);
    float*  sum = (float*)carve(HID * 4);
    float*  sq  = (float*)carve(HID * 4);

    const long long* srcI = ei;
    const long long* dstI = ei + N_EDGES;

    // ---- one-time prep ----
    hipMemsetAsync(cnt, 0, (size_t)N_NODES * 4, stream);
    k_degree<<<cdiv(N_EDGES, 256), 256, 0, stream>>>(dstI, cnt, N_EDGES);
    k_pack_x<<<cdiv((long long)N_NODES * K0PAD, 256), 256, 0, stream>>>(x, X0);

    const int dinA [4] = {NFEAT, HID, HID, HID};
    const int dinpA[4] = {K0PAD, HID, HID, HID};
    const int doutA[4] = {HID, HID, HID, DIMOUT};
    for (int i = 0; i < 4; ++i) {
        long long tot = (long long)dinpA[i] * 6 * doutA[i];
        k_pack_w<<<cdiv(tot, 256), 256, 0, stream>>>(lin_w[i], film_w[i], skip_w[i],
                                                     fskip_w[i], Bp[i], dinA[i], dinpA[i], doutA[i]);
    }

    const int mgroups = cdiv(N_NODES / 16, 8);   // 8 M-tiles per block

    // ---- 4 FiLM layers ----
    for (int i = 0; i < 4; ++i) {
        const int dout = doutA[i];
        const int nct  = 6 * dout / 16;          // 120, 120, 120, 6
        const __bf16* A = (i == 0) ? X0 : Hb;
        const int lda = dinpA[i];
        const int act = (i < 3) ? 1 : 0;

        if (i == 0) {
            const int cgroups = nct / 4;         // 30
            k_gemm<4, 1><<<mgroups * cgroups, 256, 0, stream>>>(A, lda, Bp[i], Yc,
                                                                film_b[i], nct, dout);
        } else if (i < 3) {
            const int cgroups = nct / 4;         // 30
            k_gemm<4, 10><<<mgroups * cgroups, 256, 0, stream>>>(A, lda, Bp[i], Yc,
                                                                 film_b[i], nct, dout);
        } else {
            const int cgroups = nct / 2;         // 3
            k_gemm<2, 10><<<mgroups * cgroups, 256, 0, stream>>>(A, lda, Bp[i], Yc,
                                                                 film_b[i], nct, dout);
        }

        hipMemsetAsync(agg, 0, (size_t)N_NODES * dout * 4, stream);
        k_edge<<<cdiv((long long)N_EDGES * (dout / 4), 256), 256, 0, stream>>>(
            srcI, dstI, Yc, agg, dout, act);

        float* Fdst = (i < 3) ? F : (float*)d_out;
        k_self_agg<<<cdiv((long long)N_NODES * dout, 256), 256, 0, stream>>>(
            Yc, agg, cnt, Fdst, dout, act);

        if (i < 3) {
            hipMemsetAsync(sum, 0, HID * 4, stream);
            hipMemsetAsync(sq,  0, HID * 4, stream);
            dim3 gs(cdiv(N_NODES, 512), cdiv(dout, 256));
            k_colstats<<<gs, 256, 0, stream>>>(F, sum, sq, dout);
            k_bn_apply<<<cdiv((long long)N_NODES * dout, 256), 256, 0, stream>>>(
                F, sum, sq, bn_g[i], bn_b[i], Hb, dout);
        }
    }
}
